// PointAttention2_36498632081493
// MI455X (gfx1250) — matrix operand order repaired and checked by reading, not compiled, not run on hardware
//
#include <hip/hip_runtime.h>

// ---------------------------------------------------------------------------
// PointAttention2 fused kernel for gfx1250 (MI455X, wave32, WMMA).
//
// out[b,g] = sum_o tanh(q[b,o] + be[o] + sum_h ref[bg,h]*We[o,h]) * Wv[o&31] + 8*bv
// q = query @ Wq^T + bq   (precomputed into d_ws)
//
// Main GEMM (M=262144, N=256, K=256) runs on v_wmma_f32_16x16x32_bf16 with
// fp32 accumulation; ref is streamed from HBM exactly once (256 MB ~ 11us
// floor at 23.3 TB/s). We is staged once per block into LDS as bf16 with a
// padded 528-byte row stride (conflict-free 16-lane fragment reads).
// Epilogue uses the CDNA5 hardware tanh (TRANS op, co-executes with WMMA).
// ---------------------------------------------------------------------------

typedef __attribute__((ext_vector_type(16))) __bf16 v16bf;
typedef __attribute__((ext_vector_type(8)))  __bf16 v8bf;
typedef __attribute__((ext_vector_type(8)))  float  v8f;

#define LDS_STRIDE 264  // 256 bf16 + 8 pad -> 528B rows, 16B aligned, bank-friendly

__device__ __forceinline__ __bf16 f2bf(float f) {
  return (__bf16)f;   // RTNE fptrunc; native v_cvt_*bf16* on gfx1250
}

__device__ __forceinline__ float fast_tanh(float x) {
#if __has_builtin(__builtin_amdgcn_tanhf)
  return __builtin_amdgcn_tanhf(x);           // V_TANH_F32 (gfx1250 TRANS op)
#elif __has_builtin(__builtin_amdgcn_tanh_f32)
  return __builtin_amdgcn_tanh_f32(x);
#else
  // Branch-free: tanh(x) = 1 - 2/(1 + exp(2x)); exp2 overflow/underflow
  // saturates to +-1 with no NaNs. 2*log2(e) = 2.88539008...
  float a = __builtin_amdgcn_exp2f(x * 2.885390081777927f);
  return 1.0f - 2.0f * __builtin_amdgcn_rcpf(a + 1.0f);
#endif
}

// A-fragment (16x32 bf16, this lane's half): elems 0-7 = row[kk..kk+7],
// elems 8-15 = row[kk+16..kk+23]  (ISA 7.12.2 16-bit A layout).
__device__ __forceinline__ v16bf load_a_frag(const float* __restrict__ rowp, int kk) {
  float4 x0 = *(const float4*)(rowp + kk);
  float4 x1 = *(const float4*)(rowp + kk + 4);
  float4 x2 = *(const float4*)(rowp + kk + 16);
  float4 x3 = *(const float4*)(rowp + kk + 20);
  v16bf a;
  a[0]  = f2bf(x0.x); a[1]  = f2bf(x0.y); a[2]  = f2bf(x0.z); a[3]  = f2bf(x0.w);
  a[4]  = f2bf(x1.x); a[5]  = f2bf(x1.y); a[6]  = f2bf(x1.z); a[7]  = f2bf(x1.w);
  a[8]  = f2bf(x2.x); a[9]  = f2bf(x2.y); a[10] = f2bf(x2.z); a[11] = f2bf(x2.w);
  a[12] = f2bf(x3.x); a[13] = f2bf(x3.y); a[14] = f2bf(x3.z); a[15] = f2bf(x3.w);
  return a;
}

// ---------------------------------------------------------------------------
// Kernel 1: q[b,o] = bq[o] + sum_i query[b,i] * Wq[o,i]   (B=256, 2H=512)
// ---------------------------------------------------------------------------
__global__ __launch_bounds__(256) void pa2_qdense(const float* __restrict__ query,
                                                  const float* __restrict__ Wq,
                                                  const float* __restrict__ bq,
                                                  float* __restrict__ qout) {
  __shared__ float sq[512];
  const int b = blockIdx.x;
  const int t = threadIdx.x;
  sq[t]       = query[b * 512 + t];
  sq[t + 256] = query[b * 512 + t + 256];
  __syncthreads();
  const float* wrow = Wq + (size_t)t * 512;
  float acc = 0.0f;
#pragma unroll 4
  for (int i = 0; i < 512; i += 4) {
    float4 w = *(const float4*)(wrow + i);
    acc += w.x * sq[i] + w.y * sq[i + 1] + w.z * sq[i + 2] + w.w * sq[i + 3];
  }
  qout[b * 256 + t] = acc + bq[t];
}

// ---------------------------------------------------------------------------
// Kernel 2: fused e-GEMM + tanh scorer + head reduction.
// Block = 256 threads (8 waves). Wave owns 32 M-rows (two 16x16 M-tiles)
// across the full N=256 (16 n-tiles), K=256 (8 WMMA k-steps).
// ---------------------------------------------------------------------------
__global__ __launch_bounds__(256) void pa2_main(const float* __restrict__ ref,
                                                const float* __restrict__ qbuf,
                                                const float* __restrict__ We,
                                                const float* __restrict__ be,
                                                const float* __restrict__ Wv,
                                                const float* __restrict__ bv,
                                                float* __restrict__ out) {
  __shared__ __bf16 sWe[256 * LDS_STRIDE];  // 132 KB of the 320 KB WGP LDS

  const int t = threadIdx.x;

  // Stage We (fp32, row-major [o=256][k=256]) -> bf16 LDS, coalesced reads.
#pragma unroll 4
  for (int i = 0; i < 64; ++i) {
    int idx = i * 1024 + t * 4;               // flat element index, float4 granule
    float4 w = *(const float4*)(We + idx);
    int o = idx >> 8;
    int k = idx & 255;
    __bf16* d = &sWe[o * LDS_STRIDE + k];
    d[0] = f2bf(w.x); d[1] = f2bf(w.y); d[2] = f2bf(w.z); d[3] = f2bf(w.w);
  }
  __syncthreads();

  const int wave = t >> 5;
  const int lane = t & 31;
  const int r    = lane & 15;   // row (A-M index) / column (B/C-N index)
  const int hf   = lane >> 4;   // lane half: selects K sub-chunks & C row group
  const int m_base = blockIdx.x * 256 + wave * 32;   // first of this wave's 32 rows
  const int b      = m_base >> 10;                   // G = 1024, block never crosses b
  const float* qrow = qbuf + b * 256;

  // Load both A fragments for the full K once (ref streamed exactly once).
  const float* a0p = ref + (size_t)(m_base + r) * 256;
  const float* a1p = a0p + 16 * 256;
  v16bf A0[8], A1[8];
#pragma unroll
  for (int ks = 0; ks < 8; ++ks) {
    const int kk = ks * 32 + 8 * hf;
    A0[ks] = load_a_frag(a0p, kk);
    A1[ks] = load_a_frag(a1p, kk);
  }

  // Hoist per-lane epilogue constants: qe[nt] = q[b, o] + be[o], o = nt*16 + r.
  float qe[16];
#pragma unroll
  for (int nt = 0; nt < 16; ++nt) qe[nt] = qrow[nt * 16 + r] + be[nt * 16 + r];

  const float wv0 = Wv[r];        // o & 31 == r        (even n-tiles)
  const float wv1 = Wv[16 + r];   // o & 31 == 16 + r   (odd n-tiles)

  float acc0[8] = {0.f, 0.f, 0.f, 0.f, 0.f, 0.f, 0.f, 0.f};
  float acc1[8] = {0.f, 0.f, 0.f, 0.f, 0.f, 0.f, 0.f, 0.f};

  // Fully unrolled n-tile loop: lets the scheduler hoist next k-step's
  // ds_loads past current WMMAs and overlap TRANS-epilogue with matrix ops.
#pragma unroll
  for (int nt = 0; nt < 16; ++nt) {
    const int n0 = nt * 16;
    // B-fragment row for this lane: column N = r, K chunk selected by half.
    const __bf16* wrow = &sWe[(n0 + r) * LDS_STRIDE + 8 * hf];
    v8f c0 = {};
    v8f c1 = {};
#pragma unroll
    for (int ks = 0; ks < 8; ++ks) {
      v8bf lo = *(const v8bf*)(wrow + ks * 32);
      v8bf hi = *(const v8bf*)(wrow + ks * 32 + 16);
      v16bf bf = __builtin_shufflevector(lo, hi,
                   0, 1, 2, 3, 4, 5, 6, 7, 8, 9, 10, 11, 12, 13, 14, 15);
      c0 = __builtin_amdgcn_wmma_f32_16x16x32_bf16(false, A0[ks], false, bf,
                                                   (short)0, c0, false, false);
      c1 = __builtin_amdgcn_wmma_f32_16x16x32_bf16(false, A1[ks], false, bf,
                                                   (short)0, c1, false, false);
    }
    // Fused epilogue for this n-tile: o = n0 + r for every C element on this lane.
    const float q_be = qe[nt];
    const float wv   = (nt & 1) ? wv1 : wv0;
#pragma unroll
    for (int j = 0; j < 8; ++j) {
      acc0[j] += fast_tanh(c0[j] + q_be) * wv;
      acc1[j] += fast_tanh(c1[j] + q_be) * wv;
    }
  }

  // Reduce over the 16 lanes of each half (columns N); rows: C VGPR j holds
  // M = j (lanes 0-15) or M = j + 8 (lanes 16-31).
#pragma unroll
  for (int j = 0; j < 8; ++j) {
    float v0 = acc0[j], v1 = acc1[j];
    v0 += __shfl_xor(v0, 8, 32);
    v0 += __shfl_xor(v0, 4, 32);
    v0 += __shfl_xor(v0, 2, 32);
    v0 += __shfl_xor(v0, 1, 32);
    v1 += __shfl_xor(v1, 8, 32);
    v1 += __shfl_xor(v1, 4, 32);
    v1 += __shfl_xor(v1, 2, 32);
    v1 += __shfl_xor(v1, 1, 32);
    acc0[j] = v0;
    acc1[j] = v1;
  }

  if (r == 0) {
    const float bv8 = 8.0f * bv[0];
    float* o0 = out + m_base + 8 * hf;        // rows m_base+8hf .. +7
    float* o1 = o0 + 16;                      // rows m_base+16+8hf .. +7
    float4 p;
    p.x = acc0[0] + bv8; p.y = acc0[1] + bv8; p.z = acc0[2] + bv8; p.w = acc0[3] + bv8;
    *(float4*)(o0) = p;
    p.x = acc0[4] + bv8; p.y = acc0[5] + bv8; p.z = acc0[6] + bv8; p.w = acc0[7] + bv8;
    *(float4*)(o0 + 4) = p;
    p.x = acc1[0] + bv8; p.y = acc1[1] + bv8; p.z = acc1[2] + bv8; p.w = acc1[3] + bv8;
    *(float4*)(o1) = p;
    p.x = acc1[4] + bv8; p.y = acc1[5] + bv8; p.z = acc1[6] + bv8; p.w = acc1[7] + bv8;
    *(float4*)(o1 + 4) = p;
  }
}

// ---------------------------------------------------------------------------
extern "C" void kernel_launch(void* const* d_in, const int* in_sizes, int n_in,
                              void* d_out, int out_size, void* d_ws, size_t ws_size,
                              hipStream_t stream) {
  const float* query = (const float*)d_in[0];  // [256, 512]
  const float* ref   = (const float*)d_in[1];  // [256, 1024, 256]
  const float* Wq    = (const float*)d_in[2];  // [256, 512]
  const float* bq    = (const float*)d_in[3];  // [256]
  const float* We    = (const float*)d_in[4];  // [256, 256]
  const float* be    = (const float*)d_in[5];  // [256]
  const float* Wv    = (const float*)d_in[6];  // [32]
  const float* bv    = (const float*)d_in[7];  // [1]
  // d_in[8] = num_heads (8), baked into the kernel.

  float* out  = (float*)d_out;                 // [256, 1, 1024] = 262144 floats
  float* qbuf = (float*)d_ws;                  // 65536 floats of scratch

  pa2_qdense<<<256, 256, 0, stream>>>(query, Wq, bq, qbuf);
  pa2_main<<<1024, 256, 0, stream>>>(ref, qbuf, We, be, Wv, bv, out);
}
